// ProteinFeatures_5781025980978
// MI455X (gfx1250) — matrix operand order repaired and checked by reading, not compiled, not run on hardware
//
#include <hip/hip_runtime.h>

#define B_   2
#define L_   3072
#define KNB  48
#define KDIM 416
#define NDIM 128
#define FP   424   // padded bf16 feature pitch (416 used); 424*2B = 848B rows, conflict-friendly, 16B-aligned

typedef __attribute__((ext_vector_type(8)))  float  v8f;
typedef __attribute__((ext_vector_type(8)))  __bf16 v8bf;
typedef __attribute__((ext_vector_type(16))) __bf16 v16bf;

__constant__ int c_ai[24] = {0,2,3,4,1,1,1,1,0,0,0,4,4,3,0,2,3,4,2,3,4,2,3,2};
__constant__ int c_bi[24] = {0,2,3,4,0,2,3,4,2,3,4,2,3,2,1,1,1,1,0,0,0,4,4,3};

__device__ __forceinline__ unsigned short f2bf(float f) {
  unsigned int u = __float_as_uint(f);
  u += 0x7FFFu + ((u >> 16) & 1u);      // round-to-nearest-even
  return (unsigned short)(u >> 16);
}

// ---------------- Kernel 0: atoms + bf16-transposed edge_W ----------------
__global__ __launch_bounds__(256) void prep_kernel(
    const float* __restrict__ X, const float* __restrict__ edge_W,
    float* __restrict__ atoms, unsigned short* __restrict__ Wt)
{
  int g = blockIdx.x * blockDim.x + threadIdx.x;
  if (g < B_ * L_) {
    const float* xr = X + (size_t)g * 12;
    float Nx=xr[0],Ny=xr[1],Nz=xr[2], Cax=xr[3],Cay=xr[4],Caz=xr[5];
    float Cx=xr[6],Cy=xr[7],Cz=xr[8], Ox=xr[9],Oy=xr[10],Oz=xr[11];
    float bx=Cax-Nx, by=Cay-Ny, bz=Caz-Nz;
    float cx=Cx-Cax, cy=Cy-Cay, cz=Cz-Caz;
    float ax=by*cz-bz*cy, ay=bz*cx-bx*cz, az=bx*cy-by*cx;
    float Cbx = -0.58273431f*ax + 0.56802827f*bx - 0.54067466f*cx + Cax;
    float Cby = -0.58273431f*ay + 0.56802827f*by - 0.54067466f*cy + Cay;
    float Cbz = -0.58273431f*az + 0.56802827f*bz - 0.54067466f*cz + Caz;
    float* o = atoms + (size_t)g * 15;
    o[0]=Nx;  o[1]=Ny;  o[2]=Nz;
    o[3]=Cax; o[4]=Cay; o[5]=Caz;
    o[6]=Cx;  o[7]=Cy;  o[8]=Cz;
    o[9]=Ox;  o[10]=Oy; o[11]=Oz;
    o[12]=Cbx;o[13]=Cby;o[14]=Cbz;
  } else {
    int t = g - B_ * L_;
    if (t < KDIM * NDIM) {
      int k = t / NDIM, n = t % NDIM;
      Wt[n * KDIM + k] = f2bf(edge_W[t]);   // t == k*128 + n
    }
  }
}

// ---------------- Kernel 1: per-row masked distances + top-48 ----------------
__global__ __launch_bounds__(256) void topk_kernel(
    const float* __restrict__ atoms, const float* __restrict__ mask,
    float* __restrict__ Dn, int* __restrict__ Eidx, int* __restrict__ idxOut)
{
  __shared__ float Drow[L_];
  __shared__ float wred[8];
  __shared__ int   wredi[8];
  __shared__ float sDmax;
  int row = blockIdx.x;
  int b   = row / L_;
  int tid = threadIdx.x, lane = tid & 31, wv = tid >> 5;

  float cax = atoms[(size_t)(row*5+1)*3+0];
  float cay = atoms[(size_t)(row*5+1)*3+1];
  float caz = atoms[(size_t)(row*5+1)*3+2];
  float mi  = mask[row];

  float lmax = 0.0f;
  for (int j = tid; j < L_; j += 256) {
    int rj = b*L_ + j;
    float dx = cax - atoms[(size_t)(rj*5+1)*3+0];
    float dy = cay - atoms[(size_t)(rj*5+1)*3+1];
    float dz = caz - atoms[(size_t)(rj*5+1)*3+2];
    float D  = mi * mask[rj] * sqrtf(dx*dx+dy*dy+dz*dz + 1e-6f);
    Drow[j] = D;
    lmax = fmaxf(lmax, D);
  }
  for (int off = 16; off; off >>= 1) lmax = fmaxf(lmax, __shfl_xor(lmax, off));
  if (lane == 0) wred[wv] = lmax;
  __syncthreads();
  if (tid == 0) { float m = wred[0]; for (int i=1;i<8;i++) m = fmaxf(m, wred[i]); sDmax = m; }
  __syncthreads();
  float dmax = sDmax;
  for (int j = tid; j < L_; j += 256) {
    float m2 = mi * mask[b*L_ + j];
    Drow[j] += (1.0f - m2) * dmax;
  }
  __syncthreads();

  for (int it = 0; it < KNB; ++it) {
    float bvv = 3.0e38f; int bj = L_;
    for (int j = tid; j < L_; j += 256) {
      float v = Drow[j];
      if (v < bvv || (v == bvv && j < bj)) { bvv = v; bj = j; }
    }
    for (int off = 16; off; off >>= 1) {
      float ov = __shfl_xor(bvv, off);
      int   oj = __shfl_xor(bj,  off);
      if (ov < bvv || (ov == bvv && oj < bj)) { bvv = ov; bj = oj; }
    }
    if (lane == 0) { wred[wv] = bvv; wredi[wv] = bj; }
    __syncthreads();
    if (tid == 0) {
      float fv = wred[0]; int fj = wredi[0];
      for (int i = 1; i < 8; ++i) {
        float v = wred[i]; int j2 = wredi[i];
        if (v < fv || (v == fv && j2 < fj)) { fv = v; fj = j2; }
      }
      Dn[row*KNB + it]     = fv;
      Eidx[row*KNB + it]   = fj;
      idxOut[row*KNB + it] = fj;
      Drow[fj] = 3.4e38f;
    }
    __syncthreads();
  }
}

// -------- Kernel 2: features -> bf16 WMMA GEMM (48x128 = 48x416 @ 416x128) -> LayerNorm --------
__global__ __launch_bounds__(256) void edge_kernel(
    const float* __restrict__ atoms, const int* __restrict__ Ridx,
    const int* __restrict__ chains, const float* __restrict__ posW,
    const float* __restrict__ posb, const unsigned short* __restrict__ Wt,
    const float* __restrict__ gamma, const float* __restrict__ beta,
    const float* __restrict__ Dn, const int* __restrict__ Eidx,
    float* __restrict__ Eout)
{
  __shared__ __align__(16) unsigned short feat[KNB * FP];  // 40,704 B (bf16)
  __shared__ int   nb_s[KNB];
  __shared__ float dnb_s[KNB];
  __shared__ float atomI[15];
  float* Cs = (float*)feat;                                // alias after barrier (24,576 B needed)

  int row = blockIdx.x;
  int b   = row / L_;
  int tid = threadIdx.x;

  if (tid < KNB) { nb_s[tid] = Eidx[row*KNB + tid]; dnb_s[tid] = Dn[row*KNB + tid]; }
  if (tid < 15)  atomI[tid] = atoms[(size_t)row*15 + tid];
  __syncthreads();

  // --- positional features: cols [0,16) ---
  int Ri = Ridx[row], ci = chains[row];
  for (int t = tid; t < KNB*16; t += 256) {
    int nb = t >> 4, k = t & 15;
    int rj = b*L_ + nb_s[nb];
    int dd = Ri - Ridx[rj] + 32;
    dd = dd < 0 ? 0 : (dd > 64 ? 64 : dd);
    int didx = (ci == chains[rj]) ? dd : 65;
    feat[nb*FP + k] = f2bf(posW[didx*16 + k] + posb[k]);
  }

  // --- RBF features: cols [16,416) ---
  for (int t = tid; t < KNB*25; t += 256) {
    int nb = t / 25, p = t - nb*25;
    float d;
    if (p == 0) d = dnb_s[nb];
    else {
      int pr = p - 1;
      int aI = c_ai[pr], bI = c_bi[pr];
      const float* pb = atoms + (size_t)((b*L_ + nb_s[nb])*5 + bI)*3;
      float dx = atomI[aI*3+0] - pb[0];
      float dy = atomI[aI*3+1] - pb[1];
      float dz = atomI[aI*3+2] - pb[2];
      d = sqrtf(dx*dx + dy*dy + dz*dz + 1e-6f);
    }
    int base = nb*FP + 16 + p*16;
    #pragma unroll
    for (int k = 0; k < 16; ++k) {
      float mu = 2.0f + (20.0f/15.0f) * (float)k;
      float tt = (d - mu) * 0.8f;                 // 1/sigma = 1/1.25
      feat[base + k] = f2bf(__expf(-tt*tt));
    }
  }
  __syncthreads();

  // --- WMMA GEMM: each wave owns one 16-col N-tile, three 16-row M-tiles ---
  int lane = tid & 31, h = lane >> 4, cl = lane & 15, wv = tid >> 5;
  int col  = (wv << 4) + cl;
  const unsigned short* wcol = Wt + col * KDIM;

  v8f zero = {0.f,0.f,0.f,0.f,0.f,0.f,0.f,0.f};
  v8f acc[3]; acc[0] = zero; acc[1] = zero; acc[2] = zero;

  for (int k0 = 0; k0 < KDIM; k0 += 32) {
    // B 32x16 bf16: lanes 0-15 hold K=k0..k0+15, lanes 16-31 hold K=k0+16..k0+31 (2/VGPR)
    v8bf bl = *(const v8bf*)(wcol + k0 + 16*h);
    v8bf bh = *(const v8bf*)(wcol + k0 + 16*h + 8);
    v16bf bf = __builtin_shufflevector(bl, bh, 0,1,2,3,4,5,6,7,8,9,10,11,12,13,14,15);
    #pragma unroll
    for (int mt = 0; mt < 3; ++mt) {
      // A 16x32 bf16: lane half h selects K = 16h..16h+15 (contiguous, matches B)
      const unsigned short* fr = feat + (mt*16 + cl)*FP + k0 + 16*h;
      v8bf al = *(const v8bf*)(fr);
      v8bf ah = *(const v8bf*)(fr + 8);
      v16bf af = __builtin_shufflevector(al, ah, 0,1,2,3,4,5,6,7,8,9,10,11,12,13,14,15);
      acc[mt] = __builtin_amdgcn_wmma_f32_16x16x32_bf16(
          false, af, false, bf, (short)0, acc[mt], false, false);
    }
  }

  __syncthreads();                       // all feat reads done -> safe to alias Cs
  #pragma unroll
  for (int mt = 0; mt < 3; ++mt)
    #pragma unroll
    for (int v = 0; v < 8; ++v)
      Cs[(mt*16 + 8*h + v)*NDIM + col] = acc[mt][v];   // C/D: M = vgpr + 8*(lane/16)
  __syncthreads();

  // --- LayerNorm over 128, one row per wave pass ---
  float* outRow = Eout + (size_t)row * KNB * NDIM;
  for (int r = wv; r < KNB; r += 8) {
    float x0 = Cs[r*NDIM + lane      ];
    float x1 = Cs[r*NDIM + lane + 32 ];
    float x2 = Cs[r*NDIM + lane + 64 ];
    float x3 = Cs[r*NDIM + lane + 96 ];
    float s  = x0 + x1 + x2 + x3;
    float ss = x0*x0 + x1*x1 + x2*x2 + x3*x3;
    for (int off = 16; off; off >>= 1) {
      s  += __shfl_xor(s,  off);
      ss += __shfl_xor(ss, off);
    }
    float mean = s * (1.0f/128.0f);
    float var  = ss * (1.0f/128.0f) - mean*mean;
    float rstd = rsqrtf(var + 1e-5f);
    int n0 = lane;
    outRow[r*NDIM + n0   ] = (x0-mean)*rstd*gamma[n0   ] + beta[n0   ];
    outRow[r*NDIM + n0+32] = (x1-mean)*rstd*gamma[n0+32] + beta[n0+32];
    outRow[r*NDIM + n0+64] = (x2-mean)*rstd*gamma[n0+64] + beta[n0+64];
    outRow[r*NDIM + n0+96] = (x3-mean)*rstd*gamma[n0+96] + beta[n0+96];
  }
}

extern "C" void kernel_launch(void* const* d_in, const int* in_sizes, int n_in,
                              void* d_out, int out_size, void* d_ws, size_t ws_size,
                              hipStream_t stream) {
  const float* X      = (const float*)d_in[0];
  const float* mask   = (const float*)d_in[1];
  const int*   Ridx   = (const int*)  d_in[2];
  const int*   chains = (const int*)  d_in[3];
  const float* posW   = (const float*)d_in[4];
  const float* posb   = (const float*)d_in[5];
  const float* edgeW  = (const float*)d_in[6];
  const float* gamma  = (const float*)d_in[7];
  const float* beta   = (const float*)d_in[8];

  // workspace carve-up (all offsets 16B aligned)
  float* atoms          = (float*)d_ws;                       // 92,160 f
  float* Dn             = atoms + (size_t)B_*L_*15;           // 294,912 f
  int*   Eidx           = (int*)(Dn + (size_t)B_*L_*KNB);     // 294,912 i
  unsigned short* Wt    = (unsigned short*)(Eidx + (size_t)B_*L_*KNB); // 53,248 bf16

  float* Eout   = (float*)d_out;
  int*   idxOut = (int*)d_out + (size_t)B_*L_*KNB*NDIM;       // E_idx tail of d_out

  int prepN = B_*L_ + KDIM*NDIM;
  prep_kernel<<<(prepN + 255)/256, 256, 0, stream>>>(X, edgeW, atoms, Wt);
  topk_kernel<<<B_*L_, 256, 0, stream>>>(atoms, mask, Dn, Eidx, idxOut);
  edge_kernel<<<B_*L_, 256, 0, stream>>>(atoms, Ridx, chains, posW, posb,
                                         Wt, gamma, beta, Dn, Eidx, Eout);
}
